// SimilarIntent_85332410237229
// MI455X (gfx1250) — compile-verified, matrix operand order checked by reading, not run on hardware
//
#include <hip/hip_runtime.h>
#include <math.h>

typedef _Float16 f16;
typedef __attribute__((ext_vector_type(16))) _Float16 v16h;
typedef __attribute__((ext_vector_type(8)))  _Float16 v8h;
typedef __attribute__((ext_vector_type(8)))  float    v8f;

#define NROWS 8192
#define DDIM  512
#define KCH   (DDIM/32)   // 16 k-chunks of 32 for wmma_f32_16x16x32_f16
#define TOPK  12
#define THETA 10.0f
#define STRIP 256         // columns of Sim buffered in LDS per wave per pass
#define WAVES 4
#define QCOLS (NROWS/WAVES)

// ---------------- Kernel 1: L2-normalize rows, emit fp16 ----------------
__global__ __launch_bounds__(128) void normalize_rows_kernel(
    const float* __restrict__ h, f16* __restrict__ hn)
{
  __shared__ float red[128];
  const int row = blockIdx.x;
  const int tid = threadIdx.x;
  const float* hr = h + (size_t)row * DDIM;

  float s = 0.f;
#pragma unroll
  for (int t = 0; t < DDIM/128; ++t) { float v = hr[tid + t*128]; s += v*v; }
  red[tid] = s;
  __syncthreads();
  for (int off = 64; off > 0; off >>= 1) {
    if (tid < off) red[tid] += red[tid + off];
    __syncthreads();
  }
  const float scale = 1.f / fmaxf(sqrtf(red[0]), 1e-8f);

  f16* hw = hn + (size_t)row * DDIM;
#pragma unroll
  for (int t = 0; t < DDIM/128; ++t) {
    const int d = tid + t*128;
    hw[d] = (f16)(hr[d] * scale);
  }
}

// ------- Kernel 2: fused WMMA similarity + streaming top-k + gather -------
__global__ __launch_bounds__(32*WAVES) void sim_topk_kernel(
    const float* __restrict__ h, const f16* __restrict__ hn,
    float* __restrict__ out)
{
  __shared__ float strip[WAVES][16*STRIP];        // 64 KB
  __shared__ float candv[16][WAVES*2*TOPK];       // 8 sorted lists / row
  __shared__ int   candi[16][WAVES*2*TOPK];
  __shared__ float beta[16][TOPK];
  __shared__ int   bidx[16][TOPK];

  const int tid  = threadIdx.x;
  const int wave = tid >> 5;
  const int lane = tid & 31;
  const int half = lane >> 4;   // 0: K 0..7 / 16..23 side, 1: K 8..15 / 24..31
  const int l16  = lane & 15;
  const int rowbase = blockIdx.x * 16;

  // ---- A fragments for this block's 16 rows, resident in VGPRs ----
  // 16-bit A 16x32 layout: lane(m = l16), elems 0..7 -> K = half*8+0..7,
  // elems 8..15 -> K = 16 + half*8 + 0..7.
  v16h afrag[KCH];
  {
    const f16* arow = hn + (size_t)(rowbase + l16) * DDIM;
#pragma unroll
    for (int c = 0; c < KCH; ++c) {
      const int kb = c*32 + half*8;
      v8h lo = *(const v8h*)(arow + kb);
      v8h hi = *(const v8h*)(arow + kb + 16);
#pragma unroll
      for (int e = 0; e < 8; ++e) { afrag[c][e] = lo[e]; afrag[c][8+e] = hi[e]; }
    }
  }

  // ---- per-lane streaming top-k (sorted descending) ----
  float tv[TOPK]; int ti[TOPK];
#pragma unroll
  for (int k = 0; k < TOPK; ++k) { tv[k] = -INFINITY; ti[k] = 0; }

  float* sw = &strip[wave][0];
  const int q0 = wave * QCOLS;

  for (int jb = q0; jb < q0 + QCOLS; jb += STRIP) {
    // compute a 16 x STRIP tile strip of Sim with WMMA
    for (int jt = 0; jt < STRIP/16; ++jt) {
      // B tile = rows (jb+jt*16 .. +15) of hn; same per-lane layout as A
      const f16* brow = hn + (size_t)(jb + jt*16 + l16) * DDIM;
      v8f acc = {};
#pragma unroll
      for (int c = 0; c < KCH; ++c) {
        const int kb = c*32 + half*8;
        v8h lo = *(const v8h*)(brow + kb);
        v8h hi = *(const v8h*)(brow + kb + 16);
        v16h bfrag;
#pragma unroll
        for (int e = 0; e < 8; ++e) { bfrag[e] = lo[e]; bfrag[8+e] = hi[e]; }
        acc = __builtin_amdgcn_wmma_f32_16x16x32_f16(
            /*neg_a=*/false, afrag[c], /*neg_b=*/false, bfrag,
            /*c_mod=*/(short)0, acc, /*reuse_a=*/false, /*reuse_b=*/false);
      }
      // C layout: lane(n = l16), vgpr v -> row (v + 8*half)
#pragma unroll
      for (int v = 0; v < 8; ++v)
        sw[(v + 8*half)*STRIP + jt*16 + l16] = acc[v];
    }
    __syncthreads();

    // scan: lane owns row l16, columns {half, half+2, ...} of this strip
    for (int c = half; c < STRIP; c += 2) {
      const float v = sw[l16*STRIP + c];
      if (v > tv[TOPK-1]) {
        tv[TOPK-1] = v; ti[TOPK-1] = jb + c;
#pragma unroll
        for (int s = TOPK-1; s > 0; --s) {
          if (tv[s] > tv[s-1]) {
            float fv = tv[s]; tv[s] = tv[s-1]; tv[s-1] = fv;
            int   fi = ti[s]; ti[s] = ti[s-1]; ti[s-1] = fi;
          }
        }
      }
    }
    __syncthreads();
  }

  // ---- publish the 8 sorted sublists per row ----
#pragma unroll
  for (int k = 0; k < TOPK; ++k) {
    candv[l16][(wave*2 + half)*TOPK + k] = tv[k];
    candi[l16][(wave*2 + half)*TOPK + k] = ti[k];
  }
  __syncthreads();

  // ---- 8-way sorted merge + softmax (threads 0..15, one row each) ----
  if (tid < 16) {
    const int r = tid;
    int p[2*WAVES];
#pragma unroll
    for (int l = 0; l < 2*WAVES; ++l) p[l] = 0;

    float mv[TOPK]; int mi[TOPK];
    for (int k = 0; k < TOPK; ++k) {
      float best = -INFINITY; int bl = 0;
#pragma unroll
      for (int l = 0; l < 2*WAVES; ++l) {
        if (p[l] < TOPK) {
          const float v = candv[r][l*TOPK + p[l]];
          if (v > best) { best = v; bl = l; }
        }
      }
      mv[k] = best;
      mi[k] = candi[r][bl*TOPK + p[bl]];
      p[bl]++;
    }
    const float m = mv[0];
    float e[TOPK], sum = 0.f;
#pragma unroll
    for (int k = 0; k < TOPK; ++k) { e[k] = __expf(THETA*(mv[k]-m)); sum += e[k]; }
    const float inv = 1.f / sum;
#pragma unroll
    for (int k = 0; k < TOPK; ++k) { beta[r][k] = e[k]*inv; bidx[r][k] = mi[k]; }
  }
  __syncthreads();

  // ---- out[row] = sum_k beta_k * h[idx_k]  (fp32 gather, L2-resident) ----
  for (int r = 0; r < 16; ++r) {
    float b[TOPK]; const float* src[TOPK];
#pragma unroll
    for (int k = 0; k < TOPK; ++k) {
      b[k]   = beta[r][k];
      src[k] = h + (size_t)bidx[r][k] * DDIM;
    }
#pragma unroll
    for (int t = 0; t < DDIM/(32*WAVES); ++t) {
      const int d = tid + t*(32*WAVES);
      float acc = 0.f;
#pragma unroll
      for (int k = 0; k < TOPK; ++k) acc += b[k] * src[k][d];
      out[(size_t)(rowbase + r)*DDIM + d] = acc;
    }
  }
}

extern "C" void kernel_launch(void* const* d_in, const int* in_sizes, int n_in,
                              void* d_out, int out_size, void* d_ws, size_t ws_size,
                              hipStream_t stream) {
  (void)in_sizes; (void)n_in; (void)out_size; (void)ws_size;
  const float* h  = (const float*)d_in[0];
  float*       out = (float*)d_out;
  f16*         hn  = (f16*)d_ws;             // 8192*512 fp16 = 8 MB scratch

  normalize_rows_kernel<<<NROWS, 128, 0, stream>>>(h, hn);
  sim_topk_kernel<<<NROWS/16, 32*WAVES, 0, stream>>>(h, hn, out);
}
